// LoopedMLP_41455024341306
// MI455X (gfx1250) — compile-verified, matrix operand order checked by reading, not run on hardware
//
#include <hip/hip_runtime.h>

// ---------------------------------------------------------------------------
// LoopedMLP on MI455X (gfx1250, wave32, WMMA).
// Core: bf16 WMMA GEMM (v_wmma_f32_16x16x32_bf16), 128x128 tile, 8 waves/WG,
// double-buffered LDS staged with GLOBAL_LOAD_ASYNC_TO_LDS_B128 (ASYNCcnt),
// 64-deep K chunks (16 WMMAs per barrier per wave).
// Pipeline per step: GEMM1 (mask+relu, bf16 h) -> GEMM2 (f32 x_new) ->
// pool -> control-net -> top-k mask -> select/freeze -> novelty/done update.
// ---------------------------------------------------------------------------

typedef __bf16 bf16_t;
typedef __bf16 v16bf __attribute__((ext_vector_type(16)));
typedef float  v8f   __attribute__((ext_vector_type(8)));
typedef unsigned int uint4n __attribute__((ext_vector_type(4)));

#define N_EMBD    1024
#define FULLF     12288
#define ITERSZ    4096
#define BB        8
#define TT        1024
#define ROWS_ALL  (BB*TT)        /* 8192 */
#define NOVELTY_R 0.7f

#define KC     64                /* K halves per staged chunk  */
#define PITCH  72                /* LDS row pitch in halves (64 + 8 pad) */

static __device__ __forceinline__ bf16_t f2bf(float f) {
    // round-to-nearest-even f32 -> bf16 (bit manipulation, no cvt dependence)
    unsigned u = __float_as_uint(f);
    unsigned r = (u + 0x7FFFu + ((u >> 16) & 1u)) >> 16;
    union { unsigned short s; bf16_t b; } cv;
    cv.s = (unsigned short)r;
    return cv.b;
}

// CDNA5 async copy global->LDS, 16B per lane, tracked by ASYNCcnt.
static __device__ __forceinline__ void async_b128(unsigned lds, const void* g) {
    asm volatile("global_load_async_to_lds_b128 %0, %1, off"
                 :: "v"(lds), "v"(g) : "memory");
}
static __device__ __forceinline__ void wait_async0() {
    asm volatile("s_wait_asynccnt 0x0" ::: "memory");
}

// ---------------------------------------------------------------------------
// WMMA GEMM:  C[M,N] = A[M,K] * B[N,K]^T   (A,B bf16 row-major)
// EPI==1: val = relu(val * mask[batch(row), col]) -> bf16 store (h)
// EPI==0: plain f32 store (x_new)
// Block: 256 threads = 8 waves; tile 128x128; wave = 32 rows x 64 cols
//        = 2x4 WMMA 16x16 tiles per 32-K slice; KC=64 => 16 WMMA per chunk.
// ---------------------------------------------------------------------------
union FragU { uint4n q[2]; v16bf v; };

template <int EPI>
__global__ __launch_bounds__(256) void gemm_bf16_k(
    const bf16_t* __restrict__ A, const bf16_t* __restrict__ B,
    int K, int lda, int ldb, int ldc,
    float* __restrict__ Cf, bf16_t* __restrict__ Cbf,
    const float* __restrict__ mask, int rowBase)
{
    // 128 rows x 64 K-halves, padded pitch 72 halves (144B), double buffered
    __shared__ alignas(16) bf16_t As[2][128 * PITCH];
    __shared__ alignas(16) bf16_t Bs[2][128 * PITCH];

    const int t    = threadIdx.x;
    const int lane = t & 31;
    const int wid  = t >> 5;
    const int wm   = wid & 3;    // 4 row groups of 32
    const int wn   = wid >> 2;   // 2 col groups of 64
    const int lr   = lane & 15;
    const int lh   = lane >> 4;
    const int kb   = lh * 8;     // K sub-chunk base per ISA A-layout halves

    const long long rowA0 = (long long)blockIdx.y * 128;
    const long long rowB0 = (long long)blockIdx.x * 128;

    // Staging: thread t copies 16B chunks at rows (t>>3)+{0,32,64,96},
    // half-offset (t&7)*8 within the 64-half K slice, for both A and B.
    const int srow = t >> 3;           // 0..31
    const int scol = (t & 7) << 3;     // 0,8,...,56
    const bf16_t* pA = A + (rowA0 + srow) * lda + scol;
    const bf16_t* pB = B + (rowB0 + srow) * ldb + scol;

    unsigned ldsA[2], ldsB[2];
    ldsA[0] = (unsigned)(unsigned long long)&As[0][srow * PITCH + scol];
    ldsA[1] = (unsigned)(unsigned long long)&As[1][srow * PITCH + scol];
    ldsB[0] = (unsigned)(unsigned long long)&Bs[0][srow * PITCH + scol];
    ldsB[1] = (unsigned)(unsigned long long)&Bs[1][srow * PITCH + scol];

    v8f acc[2][4];
#pragma unroll
    for (int i = 0; i < 2; ++i)
#pragma unroll
        for (int j = 0; j < 4; ++j)
            acc[i][j] = (v8f){0.f, 0.f, 0.f, 0.f, 0.f, 0.f, 0.f, 0.f};

    auto stage = [&](int nb, long long k0) {
#pragma unroll
        for (int r = 0; r < 4; ++r) {
            async_b128(ldsA[nb] + r * (32 * PITCH * 2),
                       pA + k0 + (long long)(32 * r) * lda);
            async_b128(ldsB[nb] + r * (32 * PITCH * 2),
                       pB + k0 + (long long)(32 * r) * ldb);
        }
    };

    auto compute = [&](const bf16_t* sa, const bf16_t* sb) {
#pragma unroll
        for (int ks = 0; ks < 2; ++ks) {
            const int ko = ks * 32 + kb;
            FragU fa[2], fb[4];
#pragma unroll
            for (int i = 0; i < 2; ++i) {
                int row = wm * 32 + i * 16 + lr;
                fa[i].q[0] = *reinterpret_cast<const uint4n*>(&sa[row * PITCH + ko]);
                fa[i].q[1] = *reinterpret_cast<const uint4n*>(&sa[row * PITCH + ko + 16]);
            }
#pragma unroll
            for (int j = 0; j < 4; ++j) {
                int row = wn * 64 + j * 16 + lr;
                fb[j].q[0] = *reinterpret_cast<const uint4n*>(&sb[row * PITCH + ko]);
                fb[j].q[1] = *reinterpret_cast<const uint4n*>(&sb[row * PITCH + ko + 16]);
            }
#pragma unroll
            for (int i = 0; i < 2; ++i)
#pragma unroll
                for (int j = 0; j < 4; ++j)
                    acc[i][j] = __builtin_amdgcn_wmma_f32_16x16x32_bf16(
                        false, fa[i].v, false, fb[j].v, (short)0, acc[i][j],
                        false, false);
        }
    };

    // Prologue: stage chunk 0 into buffer 0.
    stage(0, 0);
    wait_async0();
    __syncthreads();

    int buf = 0;
    const int nch = K / KC;
    for (int c = 0; c < nch - 1; ++c) {
        stage(buf ^ 1, (long long)(c + 1) * KC);   // overlaps with WMMA below
        compute(As[buf], Bs[buf]);
        wait_async0();     // our async writes into buf^1 complete
        __syncthreads();   // all waves done reading buf / writing buf^1
        buf ^= 1;
    }
    compute(As[buf], Bs[buf]);   // last chunk, no prefetch

    // Epilogue: C/D layout => VGPR v: lanes0-15 M=v, lanes16-31 M=v+8; N=lane&15
#pragma unroll
    for (int i = 0; i < 2; ++i) {
#pragma unroll
        for (int j = 0; j < 4; ++j) {
            long long row0 = rowA0 + wm * 32 + i * 16 + lh * 8;
            long long col  = rowB0 + wn * 64 + j * 16 + lr;
#pragma unroll
            for (int v = 0; v < 8; ++v) {
                long long row = row0 + v;
                float val = acc[i][j][v];
                if (EPI == 1) {
                    int bat = (int)((rowBase + row) >> 10);
                    float m = mask[(long long)bat * FULLF + col];
                    val *= m;
                    val = val > 0.f ? val : 0.f;
                    Cbf[row * ldc + col] = f2bf(val);
                } else {
                    Cf[row * ldc + col] = val;
                }
            }
        }
    }
}

// ---------------------------------------------------------------------------
// Small kernels
// ---------------------------------------------------------------------------
__global__ __launch_bounds__(256) void k_cvt(const float* __restrict__ s,
                                             bf16_t* __restrict__ d, long long n) {
    long long i = (long long)blockIdx.x * 256 + threadIdx.x;
    if (i < n) d[i] = f2bf(s[i]);
}

__global__ __launch_bounds__(256) void k_init_x(const float* __restrict__ x,
                                                float* __restrict__ xcur,
                                                bf16_t* __restrict__ xb, long long n) {
    long long i = (long long)blockIdx.x * 256 + threadIdx.x;
    if (i < n) { float v = x[i]; xcur[i] = v; xb[i] = f2bf(v); }
}

__global__ __launch_bounds__(256) void k_init_mask(float* __restrict__ active,
                                                   float* __restrict__ history,
                                                   int* __restrict__ flags) {
    int i = blockIdx.x * 256 + threadIdx.x;
    if (i < BB * FULLF) {
        float v = ((i % FULLF) < ITERSZ) ? 1.f : 0.f;
        active[i] = v; history[i] = v;
    }
    if (i == 0) flags[0] = 0;
}

__global__ __launch_bounds__(256) void k_pooled(const float* __restrict__ xnew,
                                                float* __restrict__ pooled) {
    int b = blockIdx.y;
    int c = blockIdx.x * 256 + threadIdx.x;
    float s = 0.f;
    for (int tt = 0; tt < TT; ++tt)
        s += xnew[((long long)b * TT + tt) * N_EMBD + c];
    pooled[b * N_EMBD + c] = s * (1.f / (float)TT);
}

__global__ __launch_bounds__(256) void k_ctrl1(const float* __restrict__ pooled,
                                               const float* __restrict__ Wc1,
                                               float* __restrict__ hid) {
    __shared__ float pl[N_EMBD];
    int b = blockIdx.x, t = threadIdx.x;
    for (int c = t; c < N_EMBD; c += 256) pl[c] = pooled[b * N_EMBD + c];
    __syncthreads();
    float s = 0.f;
    for (int c = 0; c < N_EMBD; ++c) s += pl[c] * Wc1[(long long)t * N_EMBD + c];
    hid[b * 256 + t] = s > 0.f ? s : 0.f;
}

__global__ __launch_bounds__(256) void k_ctrl2(const float* __restrict__ hid,
                                               const float* __restrict__ Wc2,
                                               float* __restrict__ ck) {
    __shared__ float hd[256];
    int b = blockIdx.y, t = threadIdx.x;
    hd[t] = hid[b * 256 + t];
    __syncthreads();
    int f = blockIdx.x * 256 + t;
    float s = 0.f;
    for (int d = 0; d < 256; ++d) s += hd[d] * Wc2[(long long)f * 256 + d];
    ck[(long long)b * FULLF + f] = s;
}

// Top-k (k=4096 of 12288) per row via 32-step binary search on float-ordered
// uint keys; ties resolved in ascending index order (matches lax.top_k).
__global__ __launch_bounds__(256) void k_topk(const float* __restrict__ ck,
                                              float* __restrict__ nmask) {
    __shared__ unsigned keys[FULLF];
    __shared__ unsigned red[256];
    int b = blockIdx.x, t = threadIdx.x;
    for (int f = t; f < FULLF; f += 256) {
        unsigned u = __float_as_uint(ck[(long long)b * FULLF + f]);
        keys[f] = (u & 0x80000000u) ? ~u : (u | 0x80000000u);
    }
    __syncthreads();
    unsigned lo = 0;
    for (int bit = 31; bit >= 0; --bit) {
        unsigned cand = lo | (1u << bit);
        unsigned cnt = 0;
        for (int f = t; f < FULLF; f += 256) cnt += (keys[f] >= cand) ? 1u : 0u;
        red[t] = cnt; __syncthreads();
        for (int s = 128; s > 0; s >>= 1) { if (t < s) red[t] += red[t + s]; __syncthreads(); }
        unsigned total = red[0]; __syncthreads();
        if (total >= (unsigned)ITERSZ) lo = cand;
    }
    unsigned cgt = 0;
    for (int f = t; f < FULLF; f += 256) cgt += (keys[f] > lo) ? 1u : 0u;
    red[t] = cgt; __syncthreads();
    for (int s = 128; s > 0; s >>= 1) { if (t < s) red[t] += red[t + s]; __syncthreads(); }
    unsigned gt = red[0];
    for (int f = t; f < FULLF; f += 256)
        nmask[(long long)b * FULLF + f] = (keys[f] > lo) ? 1.f : 0.f;
    __syncthreads();
    if (t == 0) {
        int rem = ITERSZ - (int)gt;
        for (int f = 0; f < FULLF && rem > 0; ++f)
            if (keys[f] == lo) { nmask[(long long)b * FULLF + f] = 1.f; --rem; }
    }
}

// x_next = done ? x_cur : x_new (in-place on x_new buffer), emit bf16 copy.
__global__ __launch_bounds__(256) void k_selectx(const float* __restrict__ xcur,
                                                 float* __restrict__ xnew,
                                                 bf16_t* __restrict__ xbnext,
                                                 const int* __restrict__ flags,
                                                 long long n) {
    long long i = (long long)blockIdx.x * 256 + threadIdx.x;
    if (i < n) {
        float v = flags[0] ? xcur[i] : xnew[i];
        xnew[i] = v;
        xbnext[i] = f2bf(v);
    }
}

// combined/history/active update + novelty + done flag (single workgroup).
__global__ __launch_bounds__(1024) void k_update(const float* __restrict__ nmask,
                                                 float* __restrict__ active,
                                                 float* __restrict__ history,
                                                 int* __restrict__ flags) {
    __shared__ float red[1024];
    int t = threadIdx.x;
    int done = flags[0];
    float s = 0.f;
    for (int i = t; i < BB * FULLF; i += 1024) {
        float h = history[i];
        float c = fminf(h + nmask[i], 1.f);
        s += c - h;
        if (!done) { history[i] = c; active[i] = nmask[i]; }
    }
    red[t] = s; __syncthreads();
    for (int st = 512; st > 0; st >>= 1) { if (t < st) red[t] += red[t + st]; __syncthreads(); }
    if (t == 0) {
        float nov = red[0] / ((float)BB * (float)ITERSZ);
        flags[0] = done || (nov < NOVELTY_R);
    }
}

__global__ __launch_bounds__(256) void k_copy(const float* __restrict__ s,
                                              float* __restrict__ d, long long n) {
    long long i = (long long)blockIdx.x * 256 + threadIdx.x;
    if (i < n) d[i] = s[i];
}

// ---------------------------------------------------------------------------
// Host launcher
// ---------------------------------------------------------------------------
extern "C" void kernel_launch(void* const* d_in, const int* in_sizes, int n_in,
                              void* d_out, int out_size, void* d_ws, size_t ws_size,
                              hipStream_t stream) {
    (void)in_sizes; (void)n_in; (void)out_size;
    const float* x   = (const float*)d_in[0];
    const float* Wm  = (const float*)d_in[1];
    const float* Wp  = (const float*)d_in[2];
    const float* Wc1 = (const float*)d_in[3];
    const float* Wc2 = (const float*)d_in[4];

    char* base = (char*)d_ws;
    size_t off = 0;
    auto alloc = [&](size_t bytes) -> void* {
        off = (off + 255) & ~(size_t)255;
        void* p = base + off; off += bytes; return p;
    };

    const long long NX = (long long)ROWS_ALL * N_EMBD;   // 8388608
    const long long NW = (long long)FULLF * N_EMBD;      // 12582912

    float*  xA   = (float*) alloc((size_t)NX * 4);
    float*  xB   = (float*) alloc((size_t)NX * 4);
    bf16_t* xbA  = (bf16_t*)alloc((size_t)NX * 2);
    bf16_t* xbB  = (bf16_t*)alloc((size_t)NX * 2);
    bf16_t* WmB  = (bf16_t*)alloc((size_t)NW * 2);
    bf16_t* WpB  = (bf16_t*)alloc((size_t)NW * 2);
    float*  pooled  = (float*)alloc((size_t)BB * N_EMBD * 4);
    float*  hid     = (float*)alloc((size_t)BB * 256 * 4);
    float*  ck      = (float*)alloc((size_t)BB * FULLF * 4);
    float*  active  = (float*)alloc((size_t)BB * FULLF * 4);
    float*  history = (float*)alloc((size_t)BB * FULLF * 4);
    float*  nmask   = (float*)alloc((size_t)BB * FULLF * 4);
    int*    flags   = (int*)  alloc(256);

    // Size the h slab (bf16, slab x 12288) to the remaining workspace.
    size_t rem = ws_size > off ? ws_size - off : 0;
    int slab = 128;
    for (int s = 8192; s >= 128; s >>= 1)
        if ((size_t)s * FULLF * 2 + 512 <= rem) { slab = s; break; }
    bf16_t* hbuf = (bf16_t*)alloc((size_t)slab * FULLF * 2);

    // ---- init (weights->bf16 once per launch; x state + base mask) ----
    k_cvt<<<(unsigned)((NW + 255) / 256), 256, 0, stream>>>(Wm, WmB, NW);
    k_cvt<<<(unsigned)((NW + 255) / 256), 256, 0, stream>>>(Wp, WpB, NW);
    k_init_x<<<(unsigned)((NX + 255) / 256), 256, 0, stream>>>(x, xA, xbA, NX);
    k_init_mask<<<(BB * FULLF + 255) / 256, 256, 0, stream>>>(active, history, flags);

    float*  xcur = xA,  *xnext = xB;
    bf16_t* xbcur = xbA, *xbnext = xbB;

    for (int step = 0; step < 3; ++step) {
        for (int r0 = 0; r0 < ROWS_ALL; r0 += slab) {
            // h = relu(active * (x @ Wm^T))  [bf16]
            gemm_bf16_k<1><<<dim3(FULLF / 128, slab / 128), 256, 0, stream>>>(
                xbcur + (long long)r0 * N_EMBD, WmB,
                N_EMBD, N_EMBD, N_EMBD, FULLF,
                nullptr, hbuf, active, r0);
            // x_new = h @ Wp^T  [f32]
            gemm_bf16_k<0><<<dim3(N_EMBD / 128, slab / 128), 256, 0, stream>>>(
                hbuf, WpB,
                FULLF, FULLF, FULLF, N_EMBD,
                xnext + (long long)r0 * N_EMBD, nullptr, nullptr, r0);
        }
        k_pooled<<<dim3(N_EMBD / 256, BB), 256, 0, stream>>>(xnext, pooled);
        k_ctrl1<<<BB, 256, 0, stream>>>(pooled, Wc1, hid);
        k_ctrl2<<<dim3(FULLF / 256, BB), 256, 0, stream>>>(hid, Wc2, ck);
        k_topk<<<BB, 256, 0, stream>>>(ck, nmask);
        k_selectx<<<(unsigned)((NX + 255) / 256), 256, 0, stream>>>(
            xcur, xnext, xbnext, flags, NX);
        k_update<<<1, 1024, 0, stream>>>(nmask, active, history, flags);

        float* tf = xcur; xcur = xnext; xnext = tf;
        bf16_t* tb = xbcur; xbcur = xbnext; xbnext = tb;
    }

    k_copy<<<(unsigned)((NX + 255) / 256), 256, 0, stream>>>(xcur, (float*)d_out, NX);
}
